// RAFT_10746008175294
// MI455X (gfx1250) — compile-verified, matrix operand order checked by reading, not run on hardware
//
#include <hip/hip_runtime.h>
#include <hip/hip_bf16.h>
#include <cstdint>
#include <cstring>

typedef _Float16 half_t;
typedef __attribute__((ext_vector_type(16))) _Float16 v16h;
typedef __attribute__((ext_vector_type(8)))  _Float16 v8h;
typedef __attribute__((ext_vector_type(8)))  float    v8f;

#define CDIV(a,b) (((a)+(b)-1)/(b))

#define BATCH 2
#define IMH 320
#define IMW 448
#define H1 40
#define W1 56
#define NPIX (H1*W1)          /* 2240 */
#define CORRC 324
#define M_CHUNK 17920

// ---------------------------------------------------------------------------
// Elementwise / utility kernels
// ---------------------------------------------------------------------------
__global__ void k_preprocess(const float* __restrict__ in, float* __restrict__ out, int n) {
  int i = blockIdx.x * blockDim.x + threadIdx.x;
  if (i < n) out[i] = in[i] * (2.0f / 255.0f) - 1.0f;
}

__global__ void k_add_relu(const float* __restrict__ a, const float* __restrict__ b,
                           float* __restrict__ dst, int n) {
  int i = blockIdx.x * blockDim.x + threadIdx.x;
  if (i < n) dst[i] = fmaxf(a[i] + b[i], 0.0f);
}

__global__ void k_axpy(const float* __restrict__ d, float* __restrict__ x, int n) {
  int i = blockIdx.x * blockDim.x + threadIdx.x;
  if (i < n) x[i] += d[i];
}

// slice channels [cOff, cOff+Cdst) of src(B,Csrc,sp) -> dst(B,Cdst,sp), op 0=copy 1=tanh 2=relu
__global__ void k_slice(const float* __restrict__ src, float* __restrict__ dst,
                        int Csrc, int cOff, int Cdst, int sp, int op) {
  int idx = blockIdx.x * blockDim.x + threadIdx.x;
  int tot = BATCH * Cdst * sp;
  if (idx >= tot) return;
  int p = idx % sp; int t = idx / sp; int c = t % Cdst; int b = t / Cdst;
  float v = src[((size_t)b * Csrc + cOff + c) * sp + p];
  if (op == 1) v = tanhf(v);
  else if (op == 2) v = fmaxf(v, 0.0f);
  dst[idx] = v;
}

// copy src(B,Cs,sp) into dst(B,Cd,sp) at channel offset off
__global__ void k_copy_ch(const float* __restrict__ src, float* __restrict__ dst,
                          int Cs, int Cd, int off, int sp) {
  int idx = blockIdx.x * blockDim.x + threadIdx.x;
  int tot = BATCH * Cs * sp;
  if (idx >= tot) return;
  int p = idx % sp; int t = idx / sp; int c = t % Cs; int b = t / Cs;
  dst[((size_t)b * Cd + off + c) * sp + p] = src[idx];
}

// write flow channels (coords1 - coords0) into encIn channels 324,325
__global__ void k_flow_ch(const float* __restrict__ coords, float* __restrict__ encIn) {
  int idx = blockIdx.x * blockDim.x + threadIdx.x;
  int tot = BATCH * 2 * NPIX;
  if (idx >= tot) return;
  int p = idx % NPIX; int t = idx / NPIX; int ch = t % 2; int b = t / 2;
  float c0 = (ch == 0) ? (float)(p % W1) : (float)(p / W1);
  encIn[((size_t)b * 326 + 324 + ch) * NPIX + p] = coords[idx] - c0;
}

__global__ void k_coords_init(float* __restrict__ coords) {
  int idx = blockIdx.x * blockDim.x + threadIdx.x;
  int tot = BATCH * 2 * NPIX;
  if (idx >= tot) return;
  int p = idx % NPIX; int ch = (idx / NPIX) % 2;
  coords[idx] = (ch == 0) ? (float)(p % W1) : (float)(p / W1);
}

// ---------------------------------------------------------------------------
// im2col (fp32 NCHW -> f16 row-major [m_cnt x KP], zero padded)
// ---------------------------------------------------------------------------
__global__ void k_im2col(const float* __restrict__ src, half_t* __restrict__ A,
                         int Cin, int H, int W, int KH, int KW, int stride, int pad,
                         int OH, int OW, int m_off, int m_cnt, int KP) {
  int idx = blockIdx.x * blockDim.x + threadIdx.x;
  if (idx >= m_cnt * KP) return;
  int r = idx / KP, k = idx - r * KP;
  int m = m_off + r;
  int ow = m % OW; int t = m / OW; int oh = t % OH; int b = t / OH;
  half_t v = (half_t)0.0f;
  int kk = Cin * KH * KW;
  if (k < kk) {
    int khw = KH * KW;
    int c = k / khw; int rem = k - c * khw; int ky = rem / KW; int kx = rem - ky * KW;
    int iy = oh * stride - pad + ky, ix = ow * stride - pad + kx;
    if (iy >= 0 && iy < H && ix >= 0 && ix < W)
      v = (half_t)src[(((size_t)b * Cin + c) * H + iy) * W + ix];
  }
  A[(size_t)r * KP + k] = v;
}

// weights (Cout,Cin*KH*KW) fp32 -> Bt f16 [CoutP x KP] (natural layout, zero pad)
__global__ void k_wgt2b(const float* __restrict__ w, half_t* __restrict__ Bt,
                        int kk, int Cout, int KP, int CoutP) {
  int idx = blockIdx.x * blockDim.x + threadIdx.x;
  if (idx >= CoutP * KP) return;
  int n = idx / KP, k = idx - n * KP;
  float v = (n < Cout && k < kk) ? w[(size_t)n * kk + k] : 0.0f;
  Bt[idx] = (half_t)v;
}

// correlation operand builders: both are [spatial][channel] transposes of NCHW f32
__global__ void k_corrT(const float* __restrict__ fb, half_t* __restrict__ T) {
  int idx = blockIdx.x * blockDim.x + threadIdx.x;           // NPIX*256
  if (idx >= NPIX * 256) return;
  int n = idx / 256, c = idx - n * 256;
  T[idx] = (half_t)fb[(size_t)c * NPIX + n];
}

// ---------------------------------------------------------------------------
// WMMA GEMM: C[m_off+m][n] = scale * sum_k A[m][k]*Bt[n][k]
// A row-major [M x KP], Bt row-major [NP x KP] (B transposed).
// One wave32 per 32x32 output tile (2x2 register tiling, 4 v_wmma per K-step).
// Per-lane fragments are two contiguous 16B loads per matrix row (CDNA5
// 16-bit A/B layout: halves h<8 at k=8*grp.., halves h>=8 at k=16+8*grp..).
// ---------------------------------------------------------------------------
__global__ void k_wmma_gemm(const half_t* __restrict__ A, const half_t* __restrict__ Bt,
                            float* __restrict__ C, int KP, int Nact, int ldC,
                            int m_off, float scale) {
  const int lane = threadIdx.x & 31;
  const int grp = lane >> 4, lr = lane & 15;
  const int tN = blockIdx.x, tM = blockIdx.y;
  const half_t* a0p = A + (size_t)(tM * 32 + lr) * KP;
  const half_t* a1p = a0p + (size_t)16 * KP;
  const half_t* b0p = Bt + (size_t)(tN * 32 + lr) * KP;
  const half_t* b1p = b0p + (size_t)16 * KP;
  const int ko0 = 8 * grp;        // halves h=0..7
  const int ko1 = 16 + 8 * grp;   // halves h=8..15
  v8f c00 = {}, c01 = {}, c10 = {}, c11 = {};
  for (int k0 = 0; k0 < KP; k0 += 32) {
    v8h a0l = *(const v8h*)(a0p + k0 + ko0);
    v8h a0h = *(const v8h*)(a0p + k0 + ko1);
    v8h a1l = *(const v8h*)(a1p + k0 + ko0);
    v8h a1h = *(const v8h*)(a1p + k0 + ko1);
    v8h b0l = *(const v8h*)(b0p + k0 + ko0);
    v8h b0h = *(const v8h*)(b0p + k0 + ko1);
    v8h b1l = *(const v8h*)(b1p + k0 + ko0);
    v8h b1h = *(const v8h*)(b1p + k0 + ko1);
    v16h a0 = __builtin_shufflevector(a0l, a0h, 0,1,2,3,4,5,6,7,8,9,10,11,12,13,14,15);
    v16h a1 = __builtin_shufflevector(a1l, a1h, 0,1,2,3,4,5,6,7,8,9,10,11,12,13,14,15);
    v16h b0 = __builtin_shufflevector(b0l, b0h, 0,1,2,3,4,5,6,7,8,9,10,11,12,13,14,15);
    v16h b1 = __builtin_shufflevector(b1l, b1h, 0,1,2,3,4,5,6,7,8,9,10,11,12,13,14,15);
    c00 = __builtin_amdgcn_wmma_f32_16x16x32_f16(false, a0, false, b0, (short)0, c00, false, false);
    c01 = __builtin_amdgcn_wmma_f32_16x16x32_f16(false, a0, false, b1, (short)0, c01, false, false);
    c10 = __builtin_amdgcn_wmma_f32_16x16x32_f16(false, a1, false, b0, (short)0, c10, false, false);
    c11 = __builtin_amdgcn_wmma_f32_16x16x32_f16(false, a1, false, b1, (short)0, c11, false, false);
  }
  const int n0 = tN * 32 + lr;        // N-subtile 0 column for this lane
  const int n1 = n0 + 16;             // N-subtile 1 column
  const int m0 = m_off + tM * 32 + 8 * grp;   // C/D layout: M = r + 8*grp
  const int m1 = m0 + 16;
  if (n0 < Nact) {
    size_t base0 = (size_t)m0 * ldC + n0;
    size_t base1 = (size_t)m1 * ldC + n0;
#pragma unroll
    for (int r = 0; r < 8; ++r) {
      C[base0 + (size_t)r * ldC] = c00[r] * scale;
      C[base1 + (size_t)r * ldC] = c10[r] * scale;
    }
  }
  if (n1 < Nact) {
    size_t base0 = (size_t)m0 * ldC + n1;
    size_t base1 = (size_t)m1 * ldC + n1;
#pragma unroll
    for (int r = 0; r < 8; ++r) {
      C[base0 + (size_t)r * ldC] = c01[r] * scale;
      C[base1 + (size_t)r * ldC] = c11[r] * scale;
    }
  }
}

// ---------------------------------------------------------------------------
// norm stats + epilogue
// ---------------------------------------------------------------------------
__global__ void k_stats(const float* __restrict__ Cm, float* __restrict__ mean,
                        float* __restrict__ rstd, int Cout, int rows, int instFlag, float eps) {
  __shared__ float s1[256], s2[256];
  int g = blockIdx.x;
  int c = instFlag ? (g % Cout) : g;
  int b = instFlag ? (g / Cout) : 0;
  const float* p = Cm + (size_t)b * rows * Cout + c;
  float a = 0.0f, q = 0.0f;
  for (int i = threadIdx.x; i < rows; i += 256) {
    float v = p[(size_t)i * Cout];
    a += v; q += v * v;
  }
  s1[threadIdx.x] = a; s2[threadIdx.x] = q;
  __syncthreads();
  for (int s = 128; s > 0; s >>= 1) {
    if (threadIdx.x < s) { s1[threadIdx.x] += s1[threadIdx.x + s]; s2[threadIdx.x] += s2[threadIdx.x + s]; }
    __syncthreads();
  }
  if (threadIdx.x == 0) {
    float m = s1[0] / rows;
    float var = s2[0] / rows - m * m;
    if (var < 0.0f) var = 0.0f;
    mean[g] = m;
    rstd[g] = rsqrtf(var + eps);
  }
}

// flags: bit0 = relu, bit1 = norm (bias cancels under norm); gamma==null -> no affine
__global__ void k_epi(const float* __restrict__ Cm, const float* __restrict__ bias,
                      const float* __restrict__ mean, const float* __restrict__ rstd,
                      const float* __restrict__ gamma, const float* __restrict__ beta,
                      float* __restrict__ out, int Cout, int OHW, int flags, int instFlag) {
  int idx = blockIdx.x * blockDim.x + threadIdx.x;
  int tot = BATCH * Cout * OHW;
  if (idx >= tot) return;
  int p = idx % OHW; int t = idx / OHW; int c = t % Cout; int b = t / Cout;
  float v = Cm[((size_t)b * OHW + p) * Cout + c];
  if (flags & 2) {
    int g = instFlag ? (b * Cout + c) : c;
    v = (v - mean[g]) * rstd[g];
    if (gamma) v = v * gamma[c] + beta[c];
  } else {
    v += bias[c];
  }
  if (flags & 1) v = fmaxf(v, 0.0f);
  out[idx] = v;
}

// ---------------------------------------------------------------------------
// pyramid / lookup / upsample
// ---------------------------------------------------------------------------
__global__ void k_avgpool(const float* __restrict__ in, float* __restrict__ out,
                          int N, int H, int W) {
  int OH = H / 2, OW = W / 2;
  int idx = blockIdx.x * blockDim.x + threadIdx.x;
  int tot = N * OH * OW;
  if (idx >= tot) return;
  int ox = idx % OW; int t = idx / OW; int oy = t % OH; int n = t / OH;
  const float* p = in + ((size_t)n * H + 2 * oy) * W + 2 * ox;
  out[idx] = 0.25f * (p[0] + p[1] + p[W] + p[W + 1]);
}

__global__ void k_corr_lookup(const float* __restrict__ p0, const float* __restrict__ p1,
                              const float* __restrict__ p2, const float* __restrict__ p3,
                              const float* __restrict__ coords, float* __restrict__ cf) {
  int idx = blockIdx.x * blockDim.x + threadIdx.x;
  int tot = BATCH * 4 * 81 * NPIX;
  if (idx >= tot) return;
  int p = idx % NPIX; int t = idx / NPIX;
  int tap = t % 81; t /= 81;
  int lvl = t % 4; int b = t / 4;
  int dy = tap / 9 - 4, dx = tap % 9 - 4;
  float cx = coords[((size_t)b * 2 + 0) * NPIX + p];
  float cy = coords[((size_t)b * 2 + 1) * NPIX + p];
  float sc = 1.0f / (float)(1 << lvl);
  float gx = cx * sc + (float)dx;
  float gy = cy * sc + (float)dy;
  int Hl = H1 >> lvl, Wl = W1 >> lvl;
  const float* img = (lvl == 0) ? p0 : (lvl == 1) ? p1 : (lvl == 2) ? p2 : p3;
  img += (size_t)(b * NPIX + p) * Hl * Wl;
  float x0 = floorf(gx), y0 = floorf(gy);
  float fx = gx - x0, fy = gy - y0;
  int ix = (int)x0, iy = (int)y0;
  auto tapf = [&](int xx, int yy) -> float {
    return (xx >= 0 && xx < Wl && yy >= 0 && yy < Hl) ? img[yy * Wl + xx] : 0.0f;
  };
  float v = tapf(ix, iy) * (1.0f - fx) * (1.0f - fy)
          + tapf(ix + 1, iy) * fx * (1.0f - fy)
          + tapf(ix, iy + 1) * (1.0f - fx) * fy
          + tapf(ix + 1, iy + 1) * fx * fy;
  int ch = lvl * 81 + (dy + 4) * 9 + (dx + 4);
  cf[((size_t)b * CORRC + ch) * NPIX + p] = v;
}

__global__ void k_upsample(const float* __restrict__ coords, float* __restrict__ out) {
  int idx = blockIdx.x * blockDim.x + threadIdx.x;
  int tot = BATCH * 2 * IMH * IMW;
  if (idx >= tot) return;
  int ox = idx % IMW; int t = idx / IMW; int oy = t % IMH; t /= IMH;
  int ch = t % 2; int b = t / 2;
  float sx = (ox + 0.5f) * 0.125f - 0.5f;
  float sy = (oy + 0.5f) * 0.125f - 0.5f;
  int x0 = (int)floorf(sx), y0 = (int)floorf(sy);
  float fx = sx - (float)x0, fy = sy - (float)y0;
  auto rd = [&](int xx, int yy) -> float {
    xx = min(max(xx, 0), W1 - 1);
    yy = min(max(yy, 0), H1 - 1);
    float c = coords[((size_t)b * 2 + ch) * NPIX + yy * W1 + xx];
    return c - ((ch == 0) ? (float)xx : (float)yy);   // flow = coords1 - coords0
  };
  float v = rd(x0, y0) * (1.0f - fx) * (1.0f - fy)
          + rd(x0 + 1, y0) * fx * (1.0f - fy)
          + rd(x0, y0 + 1) * (1.0f - fx) * fy
          + rd(x0 + 1, y0 + 1) * fx * fy;
  out[idx] = 8.0f * v;
}

// ---------------------------------------------------------------------------
// host-side orchestration
// ---------------------------------------------------------------------------
struct Scratch {
  half_t* Ah; half_t* Bh;
  float *Cb, *mean, *rstd;
  float *actA, *actB, *actC;
  float *x1p, *x2p;
  float *f1, *f2, *cnetO;
  float *pyr0, *pyr1, *pyr2, *pyr3;
  float *net, *inp, *ctx, *motion, *tmpH;
  float *corrF, *encIn, *gruIn;
  float *coords, *delta;
};

static inline int imin(int a, int b) { return a < b ? a : b; }

static void run_conv(hipStream_t st, const Scratch& S,
                     const float* src, int Cin, int H, int W,
                     const float* w, const float* bias, int Cout, int K, int stride, int pad,
                     int flags, int instFlag, const float* gamma, const float* beta,
                     float* out) {
  int OH = (H + 2 * pad - K) / stride + 1;
  int OW = (W + 2 * pad - K) / stride + 1;
  int kk = Cin * K * K;
  int KP = (kk + 31) & ~31;
  int CoutP = (Cout + 31) & ~31;   // N padded to 32 for 2x2 wave tiling
  int Mtot = BATCH * OH * OW;
  k_wgt2b<<<CDIV(CoutP * KP, 256), 256, 0, st>>>(w, S.Bh, kk, Cout, KP, CoutP);
  for (int mo = 0; mo < Mtot; mo += M_CHUNK) {
    int mc = imin(M_CHUNK, Mtot - mo);
    k_im2col<<<CDIV(mc * KP, 256), 256, 0, st>>>(src, S.Ah, Cin, H, W, K, K, stride, pad,
                                                 OH, OW, mo, mc, KP);
    dim3 g(CoutP / 32, mc / 32);
    k_wmma_gemm<<<g, 32, 0, st>>>(S.Ah, S.Bh, S.Cb, KP, Cout, Cout, mo, 1.0f);
  }
  int OHW = OH * OW;
  if (flags & 2) {
    int groups = instFlag ? BATCH * Cout : Cout;
    int rows = instFlag ? OHW : Mtot;
    k_stats<<<groups, 256, 0, st>>>(S.Cb, S.mean, S.rstd, Cout, rows, instFlag, 1e-5f);
  }
  k_epi<<<CDIV(BATCH * Cout * OHW, 256), 256, 0, st>>>(S.Cb, bias, S.mean, S.rstd,
                                                       gamma, beta, out, Cout, OHW,
                                                       flags, instFlag);
}

static void seq2_conv(hipStream_t st, const Scratch& S, const float* in, int Cin,
                      const float* w1, const float* b1, const float* w2, const float* b2,
                      int CoutF, float* out) {
  run_conv(st, S, in, Cin, H1, W1, w1, b1, 128, 3, 1, 1, 1, 0, nullptr, nullptr, S.tmpH);
  run_conv(st, S, S.tmpH, 128, H1, W1, w2, b2, CoutF, 3, 1, 1, 0, 0, nullptr, nullptr, out);
}

static void run_encoder(hipStream_t st, const Scratch& S, void* const* d_in, int base,
                        bool inst, const float* xin, float* fout) {
  auto P = [&](int i) -> const float* { return (const float*)d_in[base + i]; };
  static const int specIn[6]  = {64, 64, 64, 96, 96, 128};
  static const int specOut[6] = {64, 64, 96, 96, 128, 128};
  static const int specS[6]   = {1, 1, 2, 1, 2, 1};
  const int blkI[6] = {0, 4, 8, 14, 18, 24};    // fnet leaves, sorted-key order
  const int blkB[6] = {0, 8, 16, 28, 36, 48};   // cnet leaves, sorted-key order
  int conv1b, conv1w, conv2b, conv2w;
  const float *g1 = nullptr, *be1 = nullptr;
  if (inst) { conv1b = 28; conv1w = 29; conv2b = 30; conv2w = 31; }
  else      { conv1b = 56; conv1w = 57; conv2b = 58; conv2w = 59; be1 = P(60); g1 = P(61); }
  run_conv(st, S, xin, 3, IMH, IMW, P(conv1w), P(conv1b), 64, 7, 2, 3, 3,
           inst ? 1 : 0, g1, be1, S.actA);
  int H = 160, W = 224;
  for (int i = 0; i < 6; ++i) {
    int o = inst ? blkI[i] : blkB[i];
    int s = specS[i], cin = specIn[i], cout = specOut[i];
    int OH = (H + 2 - 3) / s + 1, OW = (W + 2 - 3) / s + 1;
    const float *n1g = nullptr, *n1b = nullptr, *n2g = nullptr, *n2b = nullptr;
    const float *dng = nullptr, *dnb = nullptr;
    int i_db = -1, i_dw = -1;
    if (inst) {
      if (s != 1) { i_db = o + 4; i_dw = o + 5; }
    } else if (s != 1) {
      i_db = o + 4; dnb = P(o + 5); dng = P(o + 6); i_dw = o + 7;
      n1b = P(o + 8); n1g = P(o + 9); n2b = P(o + 10); n2g = P(o + 11);
    } else {
      n1b = P(o + 4); n1g = P(o + 5); n2b = P(o + 6); n2g = P(o + 7);
    }
    run_conv(st, S, S.actA, cin, H, W, P(o + 1), P(o + 0), cout, 3, s, 1, 3,
             inst ? 1 : 0, n1g, n1b, S.actB);
    run_conv(st, S, S.actB, cout, OH, OW, P(o + 3), P(o + 2), cout, 3, 1, 1, 3,
             inst ? 1 : 0, n2g, n2b, S.actC);
    const float* shortcut = S.actA;
    if (s != 1) {
      if (inst)
        run_conv(st, S, S.actA, cin, H, W, P(i_dw), P(i_db), cout, 1, s, 0, 0, 0,
                 nullptr, nullptr, S.actB);
      else
        run_conv(st, S, S.actA, cin, H, W, P(i_dw), P(i_db), cout, 1, s, 0, 2, 0,
                 dng, dnb, S.actB);
      shortcut = S.actB;
    }
    int n = BATCH * cout * OH * OW;
    k_add_relu<<<CDIV(n, 256), 256, 0, st>>>(shortcut, S.actC, S.actA, n);
    H = OH; W = OW;
  }
  run_conv(st, S, S.actA, 128, H, W, P(conv2w), P(conv2b), 256, 1, 1, 0, 0, 0,
           nullptr, nullptr, fout);
}

extern "C" void kernel_launch(void* const* d_in, const int* in_sizes, int n_in,
                              void* d_out, int out_size, void* d_ws, size_t ws_size,
                              hipStream_t stream) {
  (void)in_sizes; (void)n_in; (void)out_size;
  Scratch S;
  char* base = (char*)d_ws;
  size_t off = 0;
  auto take = [&](size_t bytes) -> void* {
    void* p = base + off;
    off = (off + bytes + 255) & ~(size_t)255;
    return p;
  };
  S.Ah   = (half_t*)take(15728640ull * 2);
  S.Bh   = (half_t*)take(589824ull * 2);
  S.Cb   = (float*)take(4587520ull * 4);
  S.mean = (float*)take(512 * 4);
  S.rstd = (float*)take(512 * 4);
  S.actA = (float*)take(4587520ull * 4);
  S.actB = (float*)take(4587520ull * 4);
  S.actC = (float*)take(4587520ull * 4);
  S.x1p  = (float*)take(860160ull * 4);
  S.x2p  = (float*)take(860160ull * 4);
  S.f1   = (float*)take(1146880ull * 4);
  S.f2   = (float*)take(1146880ull * 4);
  S.cnetO= (float*)take(1146880ull * 4);
  S.pyr0 = (float*)take(10035200ull * 4);
  S.pyr1 = (float*)take(2508800ull * 4);
  S.pyr2 = (float*)take(627200ull * 4);
  S.pyr3 = (float*)take(156800ull * 4);
  S.net  = (float*)take(573440ull * 4);
  S.inp  = (float*)take(573440ull * 4);
  S.ctx  = (float*)take(573440ull * 4);
  S.motion = (float*)take(573440ull * 4);
  S.tmpH = (float*)take(573440ull * 4);
  S.corrF= (float*)take(1451520ull * 4);
  S.encIn= (float*)take(1460480ull * 4);
  S.gruIn= (float*)take(1720320ull * 4);
  S.coords = (float*)take(8960ull * 4);
  S.delta  = (float*)take(8960ull * 4);
  if (off > ws_size) return;   // workspace too small; cannot run

  const float* img1 = (const float*)d_in[0];
  const float* img2 = (const float*)d_in[1];
  auto Up = [&](int i) -> const float* { return (const float*)d_in[96 + i]; };

  // 1) preprocess
  int npre = BATCH * 3 * IMH * IMW;
  k_preprocess<<<CDIV(npre, 256), 256, 0, stream>>>(img1, S.x1p, npre);
  k_preprocess<<<CDIV(npre, 256), 256, 0, stream>>>(img2, S.x2p, npre);

  // 2) feature encoders (instance norm)
  run_encoder(stream, S, d_in, 2, true, S.x1p, S.f1);
  run_encoder(stream, S, d_in, 2, true, S.x2p, S.f2);

  // 3) correlation volume (per batch) straight into pyramid level 0
  for (int b = 0; b < BATCH; ++b) {
    k_corrT<<<CDIV(NPIX * 256, 256), 256, 0, stream>>>(S.f1 + (size_t)b * 256 * NPIX, S.Ah);
    k_corrT<<<CDIV(NPIX * 256, 256), 256, 0, stream>>>(S.f2 + (size_t)b * 256 * NPIX, S.Bh);
    dim3 g(NPIX / 32, NPIX / 32);
    k_wmma_gemm<<<g, 32, 0, stream>>>(S.Ah, S.Bh, S.pyr0 + (size_t)b * NPIX * NPIX,
                                      256, NPIX, NPIX, 0, 0.0625f);
  }
  int NQ = BATCH * NPIX;
  k_avgpool<<<CDIV(NQ * 20 * 28, 256), 256, 0, stream>>>(S.pyr0, S.pyr1, NQ, 40, 56);
  k_avgpool<<<CDIV(NQ * 10 * 14, 256), 256, 0, stream>>>(S.pyr1, S.pyr2, NQ, 20, 28);
  k_avgpool<<<CDIV(NQ * 5 * 7, 256), 256, 0, stream>>>(S.pyr2, S.pyr3, NQ, 10, 14);

  // 4) context encoder (batch norm), net/inp/ctx
  run_encoder(stream, S, d_in, 34, false, S.x1p, S.cnetO);
  int nslice = BATCH * 128 * NPIX;
  k_slice<<<CDIV(nslice, 256), 256, 0, stream>>>(S.cnetO, S.net, 256, 0, 128, NPIX, 1);
  k_slice<<<CDIV(nslice, 256), 256, 0, stream>>>(S.cnetO, S.inp, 256, 128, 128, NPIX, 2);
  seq2_conv(stream, S, S.inp, 128, Up(1), Up(0), Up(3), Up(2), 128, S.ctx);

  // 5) iterative updates
  int ncrd = BATCH * 2 * NPIX;
  k_coords_init<<<CDIV(ncrd, 256), 256, 0, stream>>>(S.coords);
  for (int it = 0; it < 4; ++it) {
    int nlook = BATCH * 4 * 81 * NPIX;
    k_corr_lookup<<<CDIV(nlook, 256), 256, 0, stream>>>(S.pyr0, S.pyr1, S.pyr2, S.pyr3,
                                                        S.coords, S.corrF);
    k_copy_ch<<<CDIV(BATCH * CORRC * NPIX, 256), 256, 0, stream>>>(S.corrF, S.encIn,
                                                                   CORRC, 326, 0, NPIX);
    k_flow_ch<<<CDIV(ncrd, 256), 256, 0, stream>>>(S.coords, S.encIn);
    seq2_conv(stream, S, S.encIn, 326, Up(5), Up(4), Up(7), Up(6), 128, S.motion);
    k_copy_ch<<<CDIV(nslice, 256), 256, 0, stream>>>(S.net, S.gruIn, 128, 384, 0, NPIX);
    k_copy_ch<<<CDIV(nslice, 256), 256, 0, stream>>>(S.ctx, S.gruIn, 128, 384, 128, NPIX);
    k_copy_ch<<<CDIV(nslice, 256), 256, 0, stream>>>(S.motion, S.gruIn, 128, 384, 256, NPIX);
    seq2_conv(stream, S, S.gruIn, 384, Up(13), Up(12), Up(15), Up(14), 128, S.net);
    seq2_conv(stream, S, S.net, 128, Up(9), Up(8), Up(11), Up(10), 2, S.delta);
    k_axpy<<<CDIV(ncrd, 256), 256, 0, stream>>>(S.delta, S.coords, ncrd);
  }

  // 6) upsample flow to full resolution
  int nout = BATCH * 2 * IMH * IMW;
  k_upsample<<<CDIV(nout, 256), 256, 0, stream>>>(S.coords, (float*)d_out);
}
